// get_model_27986006901213
// MI455X (gfx1250) — matrix operand order repaired and checked by reading, not compiled, not run on hardware
//
#include <hip/hip_runtime.h>
#include <hip/hip_bf16.h>
#include <math.h>

typedef __attribute__((ext_vector_type(16))) _Float16 v16h;
typedef __attribute__((ext_vector_type(8)))  _Float16 v8h;
typedef __attribute__((ext_vector_type(8)))  float    v8f;

#define BB   8
#define NPTS 8192
#define BN_EPS 1e-5f

// =====================================================================
// Weight prep: WT[k][n] = (half)W[n][k], zero-padded to K32 x N16.
// fp16 storage -> GEMM B-loads are per-element fp16 gathers (stride N16) with no converts.
// =====================================================================
__global__ void prep_wt(const float* __restrict__ W, _Float16* __restrict__ WT,
                        int K, int N, int K32, int N16)
{
    int g = blockIdx.x * blockDim.x + threadIdx.x;
    if (g >= K32 * N16) return;
    int k = g / N16, n = g - k * N16;
    WT[g] = (k < K && n < N) ? (_Float16)W[(size_t)n * K + k] : (_Float16)0.0f;
}

// =====================================================================
// WMMA GEMM:  Y[M,N](f32) = X[M,K](f16, row stride SX) * WT(f16, K32 x N16) + bias
// Compile-time NT in {1..4} N-tiles per wave; 4 waves/block on 4 M-tiles.
// All NT B tiles are loaded into distinct registers before the WMMA burst so
// the load clause overlaps with compute instead of a per-tile full wait.
// Requirements: M % 16 == 0, SX % 8 == 0 (16B-aligned rows), N16 % 16 == 0.
// A layout (ISA 7.12.2): lane (low,hi) holds row M=low; elems 0..7 ->
//   K=k0+hi*8+0..7 ; elems 8..15 -> K=k0+16+hi*8+0..7 (two aligned v8h loads)
// B layout: lane (low,hi) holds column N=low; elems use the same K order as A
// C/D: VGPR r, lane L: row = r + 8*(L>>4), col = L&15
// =====================================================================
template<int NT, bool EDGE>
__global__ __launch_bounds__(128)
void wmma_gemm_t(const _Float16* __restrict__ X, int SX,
                 const _Float16* __restrict__ WT,
                 const float* __restrict__ bias, float* __restrict__ Y,
                 int M, int N, int K, int N16, int ntile0)
{
    int lane = threadIdx.x & 31;
    int wave = threadIdx.x >> 5;
    int mtile = blockIdx.y * 4 + wave;
    int nt0   = ntile0 + blockIdx.x * 4;
    if (mtile * 16 >= M) return;

    int low = lane & 15, hi = lane >> 4;
    const _Float16* arow = X + (size_t)(mtile * 16 + low) * SX;
    const _Float16* brow = WT + nt0 * 16 + low;

    v8f acc[NT] = {};

    int kmain = K & ~31;
    for (int k0 = 0; k0 < kmain; k0 += 32) {
        if (k0 + 64 < K)
            __builtin_prefetch(arow + k0 + 64, 0, 1);
        v8h alo = *(const v8h*)(arow + k0 + hi * 8);
        v8h ahi = *(const v8h*)(arow + k0 + 16 + hi * 8);
        v16h a = __builtin_shufflevector(alo, ahi, 0, 1, 2, 3, 4, 5, 6, 7,
                                         8, 9, 10, 11, 12, 13, 14, 15);
        const _Float16* bk = brow + (size_t)k0 * N16;   // column nt0*16+low of WT, from row k0
        v16h bv[NT];
#pragma unroll
        for (int t = 0; t < NT; ++t)
#pragma unroll
            for (int e = 0; e < 16; ++e)
                bv[t][e] = bk[(size_t)((e < 8) ? (hi * 8 + e) : (16 + hi * 8 + (e - 8))) * N16 + t * 16];
#pragma unroll
        for (int t = 0; t < NT; ++t)
            acc[t] = __builtin_amdgcn_wmma_f32_16x16x32_f16(
                false, a, false, bv[t], (short)0, acc[t], false, false);
    }

    if (kmain < K) {                    // single guarded K-tail block
        v16h a;
#pragma unroll
        for (int e = 0; e < 16; ++e) {
            int ka = kmain + ((e < 8) ? (hi * 8 + e) : (16 + hi * 8 + (e - 8)));
            a[e] = (ka < K) ? arow[ka] : (_Float16)0.0f;
        }
        const _Float16* bk = brow + (size_t)kmain * N16;   // zero-padded rows
        v16h bv[NT];
#pragma unroll
        for (int t = 0; t < NT; ++t)
#pragma unroll
            for (int e = 0; e < 16; ++e)
                bv[t][e] = bk[(size_t)((e < 8) ? (hi * 8 + e) : (16 + hi * 8 + (e - 8))) * N16 + t * 16];
#pragma unroll
        for (int t = 0; t < NT; ++t)
            acc[t] = __builtin_amdgcn_wmma_f32_16x16x32_f16(
                false, a, false, bv[t], (short)0, acc[t], false, false);
    }

#pragma unroll
    for (int t = 0; t < NT; ++t) {
        int col = nt0 * 16 + t * 16 + low;
        if (!EDGE || col < N) {
            float bb = bias[col];
#pragma unroll
            for (int r = 0; r < 8; ++r) {
                int row = mtile * 16 + r + 8 * hi;
                Y[(size_t)row * N + col] = acc[t][r] + bb;
            }
        }
    }
}

// ---------------- transpose (B,3,N) -> (B,N,3) ----------------
__global__ void transpose_xyz(const float* __restrict__ in, float* __restrict__ out,
                              int Bn, int Nn)
{
    int g = blockIdx.x * blockDim.x + threadIdx.x;
    if (g >= Bn * Nn * 3) return;
    int b = g / (Nn * 3);
    int r = g - b * (Nn * 3);
    int n = r / 3, ch = r - n * 3;
    out[g] = in[((size_t)b * 3 + ch) * Nn + n];
}

// ---------------- farthest point sampling (one block per batch) ----------------
__global__ void fps_kernel(const float* __restrict__ xyz, float* __restrict__ mind,
                           int* __restrict__ out_idx, float* __restrict__ new_xyz,
                           int Nsrc, int npoint)
{
    __shared__ float s_val[256];
    __shared__ int   s_idx[256];
    __shared__ int   s_far;
    int b = blockIdx.x, t = threadIdx.x;
    const float* px = xyz + (size_t)b * Nsrc * 3;
    float* md = mind + (size_t)b * Nsrc;
    for (int j = t; j < Nsrc; j += 256) md[j] = 1e10f;
    if (t == 0) s_far = 0;
    __syncthreads();
    for (int i = 0; i < npoint; ++i) {
        int far = s_far;
        if (t == 0) {
            out_idx[b * npoint + i] = far;
            new_xyz[((size_t)b * npoint + i) * 3 + 0] = px[far * 3 + 0];
            new_xyz[((size_t)b * npoint + i) * 3 + 1] = px[far * 3 + 1];
            new_xyz[((size_t)b * npoint + i) * 3 + 2] = px[far * 3 + 2];
        }
        float cx = px[far * 3 + 0], cy = px[far * 3 + 1], cz = px[far * 3 + 2];
        float best = -1.0f; int bidx = 0;
        for (int j = t; j < Nsrc; j += 256) {
            float dx = px[j * 3 + 0] - cx, dy = px[j * 3 + 1] - cy, dz = px[j * 3 + 2] - cz;
            float d = dx * dx + dy * dy + dz * dz;
            float mv = fminf(md[j], d);
            md[j] = mv;
            if (mv > best) { best = mv; bidx = j; }
        }
        s_val[t] = best; s_idx[t] = bidx;
        __syncthreads();
        for (int s = 128; s > 0; s >>= 1) {
            if (t < s) {
                if (s_val[t + s] > s_val[t] ||
                    (s_val[t + s] == s_val[t] && s_idx[t + s] < s_idx[t])) {
                    s_val[t] = s_val[t + s]; s_idx[t] = s_idx[t + s];
                }
            }
            __syncthreads();
        }
        if (t == 0) s_far = s_idx[0];
        __syncthreads();
    }
}

// ---------------- ball query (one thread per center) ----------------
__global__ void ballquery_kernel(const float* __restrict__ xyz, const float* __restrict__ nxyz,
                                 int* __restrict__ idx, int Bn, int Nsrc, int S, int ns, float r2)
{
    int g = blockIdx.x * blockDim.x + threadIdx.x;
    if (g >= Bn * S) return;
    int b = g / S;
    const float* px = xyz + (size_t)b * Nsrc * 3;
    float cx = nxyz[(size_t)g * 3 + 0], cy = nxyz[(size_t)g * 3 + 1], cz = nxyz[(size_t)g * 3 + 2];
    int* o = idx + (size_t)g * ns;
    int cnt = 0, first = 0; bool found = false;
    for (int j = 0; j < Nsrc && cnt < ns; ++j) {
        float dx = px[j * 3 + 0] - cx, dy = px[j * 3 + 1] - cy, dz = px[j * 3 + 2] - cz;
        float d = dx * dx + dy * dy + dz * dz;
        if (d <= r2) {
            if (!found) { found = true; first = j; }
            o[cnt++] = j;
        }
    }
    for (int k = cnt; k < ns; ++k) o[k] = first;
}

// --- grouping: out16[b,s,k,:] = (half)concat(points[idx], xyz[idx]-center), padded stride ---
__global__ void group_kernel(const float* __restrict__ xyz, const float* __restrict__ points,
                             const float* __restrict__ nxyz, const int* __restrict__ idx,
                             _Float16* __restrict__ out, int Bn, int Nsrc, int S, int ns, int Cp,
                             int ostride)
{
    int g = blockIdx.x * blockDim.x + threadIdx.x;
    if (g >= Bn * S * ns) return;
    int b  = g / (S * ns);
    int r  = g - b * (S * ns);
    int s  = r / ns;
    int j  = idx[g];
    _Float16* o = out + (size_t)g * ostride;
    const float* pp = points + ((size_t)b * Nsrc + j) * Cp;
    for (int c = 0; c < Cp; ++c) o[c] = (_Float16)pp[c];
    size_t xb = ((size_t)b * Nsrc + j) * 3;
    size_t cb = ((size_t)b * S + s) * 3;
    o[Cp + 0] = (_Float16)(xyz[xb + 0] - nxyz[cb + 0]);
    o[Cp + 1] = (_Float16)(xyz[xb + 1] - nxyz[cb + 1]);
    o[Cp + 2] = (_Float16)(xyz[xb + 2] - nxyz[cb + 2]);
}

// ---------------- per-column sum / sumsq (one block per channel) ----------------
__global__ void colstats_kernel(const float* __restrict__ X, float* __restrict__ stats,
                                unsigned long long M, int N)
{
    __shared__ float s1[256], s2[256];
    int n = blockIdx.x, t = threadIdx.x;
    float a = 0.f, b = 0.f;
    for (unsigned long long i = t; i < M; i += 256) {
        float v = X[i * (unsigned long long)N + n];
        a += v; b += v * v;
    }
    s1[t] = a; s2[t] = b;
    __syncthreads();
    for (int s = 128; s > 0; s >>= 1) {
        if (t < s) { s1[t] += s1[t + s]; s2[t] += s2[t + s]; }
        __syncthreads();
    }
    if (t == 0) { stats[2 * n] = s1[0]; stats[2 * n + 1] = s2[0]; }
}

// ------- fused BN + ReLU: fp32 in place + fp16 copy (padded stride) for next GEMM -------
__global__ void bn_relu_kernel(float* __restrict__ X, const float* __restrict__ stats,
                               const float* __restrict__ gam, const float* __restrict__ bet,
                               unsigned long long M, int N,
                               _Float16* __restrict__ out16, int S16)
{
    unsigned long long g = (unsigned long long)blockIdx.x * blockDim.x + threadIdx.x;
    if (g >= M * (unsigned long long)N) return;
    unsigned long long row = g / (unsigned long long)N;
    int n = (int)(g - row * (unsigned long long)N);
    float mean = stats[2 * n] / (float)M;
    float var  = stats[2 * n + 1] / (float)M - mean * mean;
    float y = (X[g] - mean) * rsqrtf(var + BN_EPS) * gam[n] + bet[n];
    y = fmaxf(y, 0.0f);
    X[g] = y;
    out16[row * (unsigned long long)S16 + n] = (_Float16)y;
}

// ---------------- max over sample dim, write into strided/concat dest ----------------
__global__ void maxpool_kernel(const float* __restrict__ in, float* __restrict__ out,
                               int rows, int ns, int C, int Ctot, int c0)
{
    int g = blockIdx.x * blockDim.x + threadIdx.x;
    if (g >= rows * C) return;
    int row = g / C, c = g - row * C;
    float m = -3.0e38f;
    for (int k = 0; k < ns; ++k)
        m = fmaxf(m, in[((size_t)row * ns + k) * C + c]);
    out[(size_t)row * Ctot + c0 + c] = m;
}

// ---------- concat(xyz, points) per point (group_all input), fp16, padded stride ----------
__global__ void cat_xyz_points(const float* __restrict__ xyz, const float* __restrict__ pts,
                               _Float16* __restrict__ out, int total, int Cp, int ostride)
{
    int g = blockIdx.x * blockDim.x + threadIdx.x;
    if (g >= total) return;
    _Float16* o = out + (size_t)g * ostride;
    o[0] = (_Float16)xyz[(size_t)g * 3 + 0];
    o[1] = (_Float16)xyz[(size_t)g * 3 + 1];
    o[2] = (_Float16)xyz[(size_t)g * 3 + 2];
    const float* p = pts + (size_t)g * Cp;
    for (int c = 0; c < Cp; ++c) o[3 + c] = (_Float16)p[c];
}

// --- feature propagation: concat(points1, 3-NN interp(points2)) -> fp16, padded stride ---
__global__ void fp_interp_kernel(const float* __restrict__ xyz1, const float* __restrict__ xyz2,
                                 const float* __restrict__ p1, const float* __restrict__ p2,
                                 _Float16* __restrict__ out, int Bn, int N1, int S2, int C1, int C2,
                                 int ostride)
{
    int g = blockIdx.x * blockDim.x + threadIdx.x;
    if (g >= Bn * N1) return;
    int b = g / N1;
    _Float16* o = out + (size_t)g * ostride;
    const float* a1 = p1 + (size_t)g * C1;
    for (int c = 0; c < C1; ++c) o[c] = (_Float16)a1[c];
    const float* pb = p2 + (size_t)b * S2 * C2;
    if (S2 == 1) {
        for (int c = 0; c < C2; ++c) o[C1 + c] = (_Float16)pb[c];
        return;
    }
    float x = xyz1[(size_t)g * 3 + 0], y = xyz1[(size_t)g * 3 + 1], z = xyz1[(size_t)g * 3 + 2];
    const float* q = xyz2 + (size_t)b * S2 * 3;
    float d0 = 3e38f, d1 = 3e38f, d2 = 3e38f;
    int   i0 = 0,     i1 = 0,     i2 = 0;
    for (int j = 0; j < S2; ++j) {
        float dx = x - q[j * 3 + 0], dy = y - q[j * 3 + 1], dz = z - q[j * 3 + 2];
        float d = dx * dx + dy * dy + dz * dz;
        if (d < d0)      { d2 = d1; i2 = i1; d1 = d0; i1 = i0; d0 = d; i0 = j; }
        else if (d < d1) { d2 = d1; i2 = i1; d1 = d; i1 = j; }
        else if (d < d2) { d2 = d; i2 = j; }
    }
    float w0 = 1.0f / (d0 + 1e-8f), w1 = 1.0f / (d1 + 1e-8f), w2 = 1.0f / (d2 + 1e-8f);
    float ws = w0 + w1 + w2;
    w0 /= ws; w1 /= ws; w2 /= ws;
    const float* q0 = pb + (size_t)i0 * C2;
    const float* q1 = pb + (size_t)i1 * C2;
    const float* q2 = pb + (size_t)i2 * C2;
    for (int c = 0; c < C2; ++c)
        o[C1 + c] = (_Float16)(w0 * q0[c] + w1 * q1[c] + w2 * q2[c]);
}

// ---------------- points1 for fp1: [cls(1), xyz(3), points(3)] (fp32 side input) ----------------
__global__ void build_p1fp1(const float* __restrict__ cls, const float* __restrict__ pts,
                            float* __restrict__ out, int Bn, int Nn)
{
    int g = blockIdx.x * blockDim.x + threadIdx.x;
    if (g >= Bn * Nn) return;
    int b = g / Nn;
    float* o = out + (size_t)g * 7;
    o[0] = cls[b];
    o[1] = pts[(size_t)g * 3 + 0];
    o[2] = pts[(size_t)g * 3 + 1];
    o[3] = pts[(size_t)g * 3 + 2];
    o[4] = o[1]; o[5] = o[2]; o[6] = o[3];
}

// ---------------- generic float copy ----------------
__global__ void copyk(const float* __restrict__ in, float* __restrict__ out,
                      unsigned long long n)
{
    unsigned long long g = (unsigned long long)blockIdx.x * blockDim.x + threadIdx.x;
    if (g < n) out[g] = in[g];
}

// ---------------- log-softmax over 50 classes per row ----------------
__global__ void logsoftmax50(const float* __restrict__ in, float* __restrict__ out, int rows)
{
    int g = blockIdx.x * blockDim.x + threadIdx.x;
    if (g >= rows) return;
    const float* x = in + (size_t)g * 50;
    float mx = x[0];
    for (int o = 1; o < 50; ++o) mx = fmaxf(mx, x[o]);
    float s = 0.f;
    for (int o = 0; o < 50; ++o) s += expf(x[o] - mx);
    float l = logf(s);
    float* y = out + (size_t)g * 50;
    for (int o = 0; o < 50; ++o) y[o] = x[o] - mx - l;
}

// ---------------- l3p (8,128,256) -> (8,256,128) ----------------
__global__ void transpose_l3p(const float* __restrict__ in, float* __restrict__ out)
{
    int g = blockIdx.x * blockDim.x + threadIdx.x;
    if (g >= 8 * 256 * 128) return;
    int b = g / (256 * 128);
    int r = g - b * (256 * 128);
    int c = r / 128, n = r - c * 128;
    out[g] = in[((size_t)b * 128 + n) * 256 + c];
}

// =====================================================================

struct Layer { const float *W, *b, *g, *beta; int cin, cout; };
struct MlpOut { const float* f32; const _Float16* f16; };
static inline int rnd8(int x)  { return (x + 7)  & ~7;  }
static inline int rnd16(int x) { return (x + 15) & ~15; }
static inline int rnd32(int x) { return (x + 31) & ~31; }

extern "C" void kernel_launch(void* const* d_in, const int* in_sizes, int n_in,
                              void* d_out, int out_size, void* d_ws, size_t ws_size,
                              hipStream_t stream)
{
    (void)in_sizes; (void)n_in; (void)out_size; (void)ws_size;
    const float* xyz_in = (const float*)d_in[0];   // (8,3,8192)
    const float* cls    = (const float*)d_in[1];   // (8,1)
    int pi = 2;
    auto take = [&](int cin, int cout) -> Layer {
        Layer L;
        L.W    = (const float*)d_in[pi++];
        L.b    = (const float*)d_in[pi++];
        L.g    = (const float*)d_in[pi++];
        L.beta = (const float*)d_in[pi++];
        L.cin = cin; L.cout = cout;
        return L;
    };

    Layer sa1b0[3] = { take(6,32),  take(32,32),  take(32,64)  };
    Layer sa1b1[3] = { take(6,64),  take(64,64),  take(64,128) };
    Layer sa1b2[3] = { take(6,64),  take(64,96),  take(96,128) };
    Layer sa2b0[3] = { take(323,128), take(128,128), take(128,256) };
    Layer sa2b1[3] = { take(323,128), take(128,196), take(196,256) };
    Layer sa3b0[3] = { take(515,128), take(128,128), take(128,256) };
    Layer sa3b1[3] = { take(515,128), take(128,196), take(196,256) };
    Layer sa4[3]   = { take(515,256), take(256,512), take(512,1024) };
    Layer fp4L[2]  = { take(1536,256), take(256,256) };
    Layer fp3L[2]  = { take(768,256),  take(256,256) };
    Layer fp2L[2]  = { take(576,256),  take(256,128) };
    Layer fp1L[2]  = { take(135,128),  take(128,128) };
    Layer head1[1] = { take(128,128) };
    const float* c2W = (const float*)d_in[pi++];   // (50,128)
    const float* c2b = (const float*)d_in[pi++];   // (50,)

    // ---- workspace bump allocator (assumes ws is generously sized) ----
    char* base = (char*)d_ws;
    size_t off = 0;
    auto allocB = [&](size_t bytes) -> void* {
        void* r = (void*)(base + off);
        off += (bytes + 255) & ~(size_t)255;
        return r;
    };
    auto allocF = [&](size_t n) -> float*     { return (float*)allocB(n * 4); };
    auto allocH = [&](size_t n) -> _Float16*  { return (_Float16*)allocB(n * 2); };

    float*    pts     = allocF((size_t)BB * NPTS * 3);
    float*    mind    = allocF((size_t)BB * NPTS);
    int*      fps_i   = (int*)allocB((size_t)BB * 512 * 4);
    int*      grp_idx = (int*)allocB((size_t)BB * 512 * 128 * 4);
    float*    l1_xyz  = allocF((size_t)BB * 512 * 3);
    float*    l2_xyz  = allocF((size_t)BB * 256 * 3);
    float*    l3_xyz  = allocF((size_t)BB * 128 * 3);
    float*    l1_pts  = allocF((size_t)BB * 512 * 320);
    float*    l2_pts  = allocF((size_t)BB * 256 * 512);
    float*    l3_pts  = allocF((size_t)BB * 128 * 512);
    float*    l4_pts  = allocF((size_t)BB * 1024);
    float*    l3p     = allocF((size_t)BB * 128 * 256);
    float*    l2p     = allocF((size_t)BB * 256 * 256);
    float*    l1p     = allocF((size_t)BB * 512 * 128);
    float*    p1f     = allocF((size_t)BB * NPTS * 7);
    float*    stats   = allocF(2048);
    _Float16* WT      = allocH((size_t)1536 * 1024);   // max K32*N16
    _Float16* GRP16   = allocH((size_t)262144 * 328);  // max first-layer fp16 input
    float*    ACT_A   = allocF((size_t)524288 * 128);  // fp32 pre/post-BN activations
    float*    ACT_B   = allocF((size_t)524288 * 128);
    _Float16* A16_A   = allocH((size_t)524288 * 128);  // fp16 copies for next GEMM
    _Float16* A16_B   = allocH((size_t)524288 * 128);

    auto launch_gemm = [&](const _Float16* X, int SX, const float* W, const float* bias,
                           float* Y, size_t M, int N, int K) {
        int K32 = rnd32(K), N16 = rnd16(N);
        int wtot = K32 * N16;
        prep_wt<<<(wtot + 255) / 256, 256, 0, stream>>>(W, WT, K, N, K32, N16);
        unsigned gy = (unsigned)((M + 63) / 64);
        int nfull = N / 64;
        if (nfull > 0) {
            wmma_gemm_t<4, false><<<dim3((unsigned)nfull, gy), 128, 0, stream>>>(
                X, SX, WT, bias, Y, (int)M, N, K, N16, 0);
        }
        int rem = N - nfull * 64;
        if (rem > 0) {
            dim3 g1(1, gy);
            int nt0 = nfull * 4;
            switch ((rem + 15) / 16) {
            case 1: wmma_gemm_t<1, true><<<g1, 128, 0, stream>>>(X, SX, WT, bias, Y, (int)M, N, K, N16, nt0); break;
            case 2: wmma_gemm_t<2, true><<<g1, 128, 0, stream>>>(X, SX, WT, bias, Y, (int)M, N, K, N16, nt0); break;
            case 3: wmma_gemm_t<3, true><<<g1, 128, 0, stream>>>(X, SX, WT, bias, Y, (int)M, N, K, N16, nt0); break;
            default: wmma_gemm_t<4, true><<<g1, 128, 0, stream>>>(X, SX, WT, bias, Y, (int)M, N, K, N16, nt0); break;
            }
        }
    };

    auto run_mlp = [&](const _Float16* in16, int SX0, size_t M,
                       const Layer* Ls, int nl) -> MlpOut {
        const _Float16* cur = in16;
        float* curF = nullptr;
        int SX = SX0;
        for (int l = 0; l < nl; ++l) {
            float*    dstF = (l & 1) ? ACT_B : ACT_A;
            _Float16* dstH = (l & 1) ? A16_B : A16_A;
            int Nc = Ls[l].cout, Kc = Ls[l].cin, S16 = rnd8(Nc);
            launch_gemm(cur, SX, Ls[l].W, Ls[l].b, dstF, M, Nc, Kc);
            colstats_kernel<<<Nc, 256, 0, stream>>>(dstF, stats, (unsigned long long)M, Nc);
            unsigned long long tot = (unsigned long long)M * Nc;
            bn_relu_kernel<<<(unsigned)((tot + 255) / 256), 256, 0, stream>>>(
                dstF, stats, Ls[l].g, Ls[l].beta, (unsigned long long)M, Nc, dstH, S16);
            cur = dstH; curF = dstF;
            SX = S16;
        }
        return { curF, cur };
    };

    auto run_sa_branch = [&](const float* sxyz, const float* spoints, int Nsrc, int Cp,
                             const float* nxyz, int S, float radius, int ns,
                             const Layer* Ls, int nl, float* outPts, int Ctot, int c0) {
        int centers = BB * S;
        ballquery_kernel<<<(centers + 255) / 256, 256, 0, stream>>>(
            sxyz, nxyz, grp_idx, BB, Nsrc, S, ns, radius * radius);
        int tot = BB * S * ns;
        int SX0 = rnd8(Cp + 3);
        group_kernel<<<(tot + 255) / 256, 256, 0, stream>>>(
            sxyz, spoints, nxyz, grp_idx, GRP16, BB, Nsrc, S, ns, Cp, SX0);
        MlpOut r = run_mlp(GRP16, SX0, (size_t)tot, Ls, nl);
        int Cl = Ls[nl - 1].cout;
        int tp = BB * S * Cl;
        maxpool_kernel<<<(tp + 255) / 256, 256, 0, stream>>>(
            r.f32, outPts, BB * S, ns, Cl, Ctot, c0);
    };

    auto run_fp = [&](const float* xyz1, int N1, const float* xyz2, int S2,
                      const float* p1, int C1, const float* p2, int C2,
                      const Layer* Ls, int nl, float* outBuf) -> MlpOut {
        int tot = BB * N1;
        int SX0 = rnd8(C1 + C2);
        fp_interp_kernel<<<(tot + 255) / 256, 256, 0, stream>>>(
            xyz1, xyz2, p1, p2, GRP16, BB, N1, S2, C1, C2, SX0);
        MlpOut r = run_mlp(GRP16, SX0, (size_t)tot, Ls, nl);
        if (outBuf) {
            int Cl = Ls[nl - 1].cout;
            unsigned long long n = (unsigned long long)tot * Cl;
            copyk<<<(unsigned)((n + 255) / 256), 256, 0, stream>>>(r.f32, outBuf, n);
        }
        return r;
    };

    // ---------------- pipeline ----------------
    transpose_xyz<<<(BB * NPTS * 3 + 255) / 256, 256, 0, stream>>>(xyz_in, pts, BB, NPTS);

    // SA1 (N=8192 -> 512 centers, 3 radii)
    fps_kernel<<<BB, 256, 0, stream>>>(pts, mind, fps_i, l1_xyz, NPTS, 512);
    run_sa_branch(pts, pts, NPTS, 3, l1_xyz, 512, 0.03f,  32, sa1b0, 3, l1_pts, 320, 0);
    run_sa_branch(pts, pts, NPTS, 3, l1_xyz, 512, 0.06f,  64, sa1b1, 3, l1_pts, 320, 64);
    run_sa_branch(pts, pts, NPTS, 3, l1_xyz, 512, 0.12f, 128, sa1b2, 3, l1_pts, 320, 192);

    // SA2 (512 -> 256 centers, 2 radii)
    fps_kernel<<<BB, 256, 0, stream>>>(l1_xyz, mind, fps_i, l2_xyz, 512, 256);
    run_sa_branch(l1_xyz, l1_pts, 512, 320, l2_xyz, 256, 0.12f,  64, sa2b0, 3, l2_pts, 512, 0);
    run_sa_branch(l1_xyz, l1_pts, 512, 320, l2_xyz, 256, 0.24f, 128, sa2b1, 3, l2_pts, 512, 256);

    // SA3 (256 -> 128 centers, 2 radii)
    fps_kernel<<<BB, 256, 0, stream>>>(l2_xyz, mind, fps_i, l3_xyz, 256, 128);
    run_sa_branch(l2_xyz, l2_pts, 256, 512, l3_xyz, 128, 0.24f, 32, sa3b0, 3, l3_pts, 512, 0);
    run_sa_branch(l2_xyz, l2_pts, 256, 512, l3_xyz, 128, 0.48f, 64, sa3b1, 3, l3_pts, 512, 256);

    // SA4 group-all: concat(l3_xyz, l3_pts) -> MLP -> max over the 128 pts per batch
    {
        int tot = BB * 128;
        cat_xyz_points<<<(tot + 255) / 256, 256, 0, stream>>>(l3_xyz, l3_pts, GRP16, tot, 512, 520);
        MlpOut r = run_mlp(GRP16, 520, (size_t)tot, sa4, 3);
        int tp = BB * 1024;
        maxpool_kernel<<<(tp + 255) / 256, 256, 0, stream>>>(r.f32, l4_pts, BB, 128, 1024, 1024, 0);
    }

    // Feature propagation
    run_fp(l3_xyz, 128, l3_xyz,   1, l3_pts, 512, l4_pts, 1024, fp4L, 2, l3p);
    run_fp(l2_xyz, 256, l3_xyz, 128, l2_pts, 512, l3p,     256, fp3L, 2, l2p);
    run_fp(l1_xyz, 512, l2_xyz, 256, l1_pts, 320, l2p,     256, fp2L, 2, l1p);
    build_p1fp1<<<(BB * NPTS + 255) / 256, 256, 0, stream>>>(cls, pts, p1f, BB, NPTS);
    MlpOut l0 = run_fp(pts, NPTS, l1_xyz, 512, p1f, 7, l1p, 128, fp1L, 2, nullptr);

    // Head + classifier + log-softmax (fp1's fp16 activations feed head directly)
    {
        size_t M = (size_t)BB * NPTS;
        MlpOut feat = run_mlp(l0.f16, 128, M, head1, 1);
        launch_gemm(feat.f16, 128, c2W, c2b, ACT_B, M, 50, 128);
        logsoftmax50<<<(unsigned)((M + 255) / 256), 256, 0, stream>>>(ACT_B, (float*)d_out, (int)M);
    }

    // second output: transpose(l3p)
    transpose_l3p<<<(8 * 256 * 128 + 255) / 256, 256, 0, stream>>>(
        l3p, (float*)d_out + (size_t)BB * NPTS * 50);
}
